// infoNCELoss_16466904613067
// MI455X (gfx1250) — matrix operand order repaired and checked by reading, not compiled, not run on hardware
//
#include <hip/hip_runtime.h>
#include <hip/hip_bf16.h>
#include <math.h>

// ---------------------------------------------------------------------------
// InfoNCE / NT-Xent loss, fused for MI455X (gfx1250, wave32, WMMA + TDM).
//   z1,z2: (4096,256) f32  ->  scalar loss
// Pipeline: normalize->f16 | TDM-staged WMMA Gram + exp + rowsum/pos | reduce
// ---------------------------------------------------------------------------

typedef __attribute__((ext_vector_type(8)))  _Float16     v8h;
typedef __attribute__((ext_vector_type(16))) _Float16     v16h;
typedef __attribute__((ext_vector_type(8)))  float        v8f;
typedef __attribute__((ext_vector_type(4)))  unsigned int u32x4;
typedef __attribute__((ext_vector_type(4)))  int          i32x4;
typedef __attribute__((ext_vector_type(8)))  int          i32x8;

#define D             256    // feature dim (reference fixed)
#define TILE          16     // WMMA M/N
#define KSTEP         32     // WMMA K for f16
#define NT_PER_WAVE   8      // 8 * 16 = 128 output cols per wave
#define WAVES_PER_BLK 8      // 8 * 16 = 128 output rows per block
#define NCOLS         (NT_PER_WAVE * TILE)          // 128
#define PROW          40     // padded LDS row stride in f16 (80B: bank-spread, 16B aligned)
#define BUFB          (NCOLS * PROW * 2)            // 10240 bytes per LDS buffer
#define NSTEPS        (D / KSTEP)                   // 8

// -------------------------- row normalization ------------------------------
__global__ void __launch_bounds__(256)
norm_kernel(const float* __restrict__ z1, const float* __restrict__ z2,
            _Float16* __restrict__ zn, int bsz) {
    int gwave = (blockIdx.x * blockDim.x + threadIdx.x) >> 5;
    int lane  = threadIdx.x & 31;
    int N = 2 * bsz;
    if (gwave >= N) return;
    const float* src = (gwave < bsz) ? (z1 + (size_t)gwave * D)
                                     : (z2 + (size_t)(gwave - bsz) * D);
    float v[8];
    float ss = 0.f;
#pragma unroll
    for (int i = 0; i < 8; ++i) {
        v[i] = src[lane * 8 + i];
        ss  += v[i] * v[i];
    }
#pragma unroll
    for (int m = 16; m >= 1; m >>= 1) ss += __shfl_xor(ss, m, 32);
    float inv = 1.0f / fmaxf(sqrtf(ss), 1e-8f);   // torch CosineSimilarity eps
    v8h o;
#pragma unroll
    for (int i = 0; i < 8; ++i) o[i] = (_Float16)(v[i] * inv);
    *(v8h*)(zn + (size_t)gwave * D + lane * 8) = o;
}

// ------------------------------ zero scratch -------------------------------
__global__ void zero_kernel(float* __restrict__ p, int n) {
    int i = blockIdx.x * blockDim.x + threadIdx.x;
    if (i < n) p[i] = 0.f;
}

// ---------------- TDM: stage one 32(K) x 128(N) f16 B panel into LDS -------
// D# per CDNA5 ISA: group0 = {flags, lds_addr, gaddr_lo, gaddr_hi|type=2},
// group1 packs data_size/pad + tensor dims 256 x N + tile 32 x 128 + stride.
// Pad: every 16 DWORDs (64B row) insert 4 DWORDs -> 80B LDS row stride.
__device__ __forceinline__ void tdm_load_bpanel(const _Float16* zn, int n0,
                                                int k0, unsigned lds_off, int N) {
    unsigned long long ga = (unsigned long long)(uintptr_t)zn
                          + ((unsigned long long)n0 * D + (unsigned)k0) * 2ull;
    u32x4 g0;
    g0[0] = 1u;                                   // count=1, user descriptor
    g0[1] = lds_off;                              // LDS byte address
    g0[2] = (unsigned)ga;                         // global addr [31:0]
    g0[3] = (unsigned)(ga >> 32) | (2u << 30);    // global addr [56:32] | type=2
    i32x8 g1;
    g1[0] = (int)((1u << 16)      // data_size = 2 bytes
                | (1u << 20)      // pad_enable
                | (3u << 22)      // pad_interval code 3 = 16 DWORDs (64B)
                | (3u << 25));    // pad_amount  code 3 = 4 DWORDs (16B)
    g1[1] = (int)((unsigned)D << 16);        // tensor_dim0[15:0] = 256
    g1[2] = (int)((unsigned)N << 16);        // dim0 hi=0 | tensor_dim1[15:0]
    g1[3] = (int)((unsigned)KSTEP << 16);    // dim1 hi=0 | tile_dim0 = 32
    g1[4] = NCOLS;                           // tile_dim1 = 128, tile_dim2 = 0
    g1[5] = D;                               // tensor_dim0_stride = 256
    g1[6] = 0;
    g1[7] = 0;
    i32x4 gz = {0, 0, 0, 0};
#if defined(__clang_major__) && __clang_major__ >= 23
    i32x8 gz8 = {0, 0, 0, 0, 0, 0, 0, 0};
    __builtin_amdgcn_tensor_load_to_lds(g0, g1, gz, gz, gz8, 0);
#else
    __builtin_amdgcn_tensor_load_to_lds(g0, g1, gz, gz, 0);
#endif
}

// --------------- fused Gram-matrix / exp / row-sum / pos-capture -----------
__global__ void __launch_bounds__(256)
simtile_kernel(const _Float16* __restrict__ zn,
               float* __restrict__ rowsum,    // (2B) row sums excl. diagonal
               float* __restrict__ pos,       // (2B) exp_s[R][partner(R)]
               int bsz, float invT) {
    __shared__ __attribute__((aligned(16))) _Float16 bpanel[2][NCOLS * PROW];

    const int lane = threadIdx.x & 31;
    const int wave = threadIdx.x >> 5;
    const int lidx = lane & 15;       // local col / local A-row
    const int half = lane >> 4;       // K-half selector per WMMA layout
    const int N    = 2 * bsz;

    const int m0 = blockIdx.y * (WAVES_PER_BLK * TILE) + wave * TILE;
    const int n0 = blockIdx.x * NCOLS;

    v8f acc[NT_PER_WAVE] = {};
    const _Float16* arow = zn + (size_t)(m0 + lidx) * D;

    if (wave == 0) tdm_load_bpanel(zn, n0, 0, 0u, N);        // prime buffer 0

#pragma unroll
    for (int i = 0; i < NSTEPS; ++i) {
        const int k0 = i * KSTEP;
        if (wave == 0) {
            if (i + 1 < NSTEPS) {
                tdm_load_bpanel(zn, n0, k0 + KSTEP,
                                (unsigned)(((i + 1) & 1) * BUFB), N);
                __builtin_amdgcn_s_wait_tensorcnt((short)1);  // buffer i landed
            } else {
                __builtin_amdgcn_s_wait_tensorcnt((short)0);
            }
        }
        __syncthreads();
        // TDM wrote bpanel behind the compiler's back: escape its address into
        // a memory-clobbering asm so the LDS loads below cannot fold to undef.
        asm volatile("" :: "v"(&bpanel[0][0]) : "memory");

        if (k0 + KSTEP < D)
            __builtin_prefetch(arow + k0 + KSTEP, 0, 3);      // near-scope prefetch

        // A operand: lane<16 -> row m0+lidx K {k0..+7, k0+16..+23};
        //            lane>=16 -> same row K {k0+8..+15, k0+24..+31}.
        v8h a0 = *(const v8h*)(arow + k0 + half * 8);
        v8h a1 = *(const v8h*)(arow + k0 + 16 + half * 8);
        v16h a = __builtin_shufflevector(a0, a1,
                    0,1,2,3,4,5,6,7,8,9,10,11,12,13,14,15);

        const _Float16* bbuf = bpanel[i & 1];
#pragma unroll
        for (int t = 0; t < NT_PER_WAVE; ++t) {
            // B operand from LDS: lane holds column n (= one staged Zn row).
            const _Float16* brow = bbuf + (t * TILE + lidx) * PROW + half * 8;
            v8h b0 = *(const v8h*)(brow);
            v8h b1 = *(const v8h*)(brow + 16);
            v16h b = __builtin_shufflevector(b0, b1,
                        0,1,2,3,4,5,6,7,8,9,10,11,12,13,14,15);

            acc[t] = __builtin_amdgcn_wmma_f32_16x16x32_f16(
                         false, a, false, b, (short)0, acc[t], false, false);
        }
        __syncthreads();                  // protect buffer (i&1) before refill
    }

    // Epilogue: exp(S * invT), mask diagonal, capture +-bsz diagonals, rowsum.
    // C/D layout: lane<16 VGPR v -> (M=v, N=lidx); lane>=16 -> (M=v+8, N=lidx).
    float rs[8];
#pragma unroll
    for (int v = 0; v < 8; ++v) rs[v] = 0.f;

#pragma unroll
    for (int t = 0; t < NT_PER_WAVE; ++t) {
        int Cg = n0 + t * TILE + lidx;
#pragma unroll
        for (int v = 0; v < 8; ++v) {
            int R = m0 + v + 8 * half;
            float e = __expf(acc[t][v] * invT);
            if (Cg != R) rs[v] += e;                 // exclude self-similarity
            int partner = (R < bsz) ? (R + bsz) : (R - bsz);
            if (Cg == partner) pos[R] = e;           // unique writer across grid
        }
    }

#pragma unroll
    for (int v = 0; v < 8; ++v) {
        float s = rs[v];
        s += __shfl_xor(s, 1, 32);
        s += __shfl_xor(s, 2, 32);
        s += __shfl_xor(s, 4, 32);
        s += __shfl_xor(s, 8, 32);
        if (lidx == 0) atomicAdd(&rowsum[m0 + v + 8 * half], s);
    }
}

// ------------------------------ final reduce -------------------------------
__global__ void __launch_bounds__(256)
loss_kernel(const float* __restrict__ rowsum, const float* __restrict__ pos,
            float* __restrict__ out, int bsz) {
    __shared__ float sh[256];
    int tid = threadIdx.x;
    int N = 2 * bsz;
    float acc = 0.f;
    for (int r = tid; r < N; r += 256)
        acc += logf(rowsum[r]) - logf(pos[r]);   // -log(pos/rowsum)
    sh[tid] = acc;
    __syncthreads();
    for (int s = 128; s > 0; s >>= 1) {
        if (tid < s) sh[tid] += sh[tid + s];
        __syncthreads();
    }
    if (tid == 0) out[0] = 0.5f * (float)bsz * sh[0];
}

// ------------------------------- launcher ----------------------------------
extern "C" void kernel_launch(void* const* d_in, const int* in_sizes, int n_in,
                              void* d_out, int out_size, void* d_ws, size_t ws_size,
                              hipStream_t stream) {
    const float* z1 = (const float*)d_in[0];
    const float* z2 = (const float*)d_in[1];
    float* out = (float*)d_out;

    const int bsz = in_sizes[0] / D;   // 4096
    const int N   = 2 * bsz;           // 8192
    const float invT = 1.0f / sqrtf((float)bsz);

    // Workspace layout: [ zn f16 (N*D) | rowsum f32 (N) | pos f32 (N) ]
    _Float16* zn  = (_Float16*)d_ws;
    float* rowsum = (float*)((char*)d_ws + (size_t)N * D * sizeof(_Float16));
    float* pos    = rowsum + N;

    norm_kernel<<<(N + WAVES_PER_BLK - 1) / WAVES_PER_BLK, 256, 0, stream>>>(
        z1, z2, zn, bsz);
    zero_kernel<<<(N + 255) / 256, 256, 0, stream>>>(rowsum, N);

    dim3 grid(N / NCOLS,                    // 64 column strips
              N / (WAVES_PER_BLK * TILE));  // 64 row groups
    simtile_kernel<<<grid, 256, 0, stream>>>(zn, rowsum, pos, bsz, invT);

    loss_kernel<<<1, 256, 0, stream>>>(rowsum, pos, out, bsz);
}
